// Clustered_Attention_Chunking_21973052686898
// MI455X (gfx1250) — matrix operand order repaired and checked by reading, not compiled, not run on hardware
//
#include <hip/hip_runtime.h>

typedef __attribute__((ext_vector_type(16))) __bf16 v16bf;
typedef __attribute__((ext_vector_type(8)))  float  v8f;

#define N_BATCH 2048
#define C_DIM   64
#define E_DIM   512
#define NHEADS  8
#define DH      64
#define CE      (C_DIM * E_DIM)   // 32768
#define WELEMS  (E_DIM * E_DIM)   // 262144 elements per weight matrix

// ---------------------------------------------------------------------------
// WMMA wrapper: D = A(16x32 bf16) * B(32x16 bf16) + C(16x16 f32)
// ---------------------------------------------------------------------------
__device__ __forceinline__ v8f wmma_bf16(v16bf a, v16bf b, v8f c) {
  return __builtin_amdgcn_wmma_f32_16x16x32_bf16(false, a, false, b,
                                                 (short)0, c, false, false);
}

union bfcast { uint4 u[2]; v16bf v; };

// ---------------------------------------------------------------------------
// A-operand (16x32 bf16), contiguous-k order used by this code (B loads match):
//   lane l<16 : M=l,    K = {0..15}; lane l>=16: M=l-16, K = {16..31}
// Two 16B loads per lane (klo is 16B aligned for all strides used here).
// ---------------------------------------------------------------------------
__device__ __forceinline__ v16bf load_A_lds(const __bf16* base, int stride, int lane) {
  int m   = lane & 15;
  int klo = (lane < 16) ? 0 : 16;
  const __bf16* p = base + m * stride + klo;
  bfcast cv;
  cv.u[0] = *(const uint4*)p;          // K klo..klo+7
  cv.u[1] = *(const uint4*)(p + 8);    // K klo+8..klo+15
  return cv.v;
}

// ---------------------------------------------------------------------------
// B-operand (32x16 bf16): lane holds column n = lane&15, K rows (lane>>4)*16..+15.
// B[k][n] = W[nbase+n][kbase+k]  -> 16 contiguous bf16 per lane (two 16B loads).
// ---------------------------------------------------------------------------
__device__ __forceinline__ v16bf load_B_wbf(const __bf16* W, int nbase, int kbase,
                                            int lane) {
  int nn = lane & 15;
  int kg = (lane >> 4) << 4;
  const __bf16* p = W + (size_t)(nbase + nn) * E_DIM + kbase + kg;
  bfcast cv;
  cv.u[0] = *(const uint4*)p;
  cv.u[1] = *(const uint4*)(p + 8);
  return cv.v;
}

// B[k][n] = base[n*stride + k] from LDS (K natural / V^T layouts)
__device__ __forceinline__ v16bf load_B_lds(const __bf16* base, int stride, int lane) {
  int nn = lane & 15;
  int kg = (lane >> 4) << 4;
  const __bf16* p = base + nn * stride + kg;
  bfcast cv;
  cv.u[0] = *(const uint4*)p;
  cv.u[1] = *(const uint4*)(p + 8);
  return cv.v;
}

// ---------------------------------------------------------------------------
// One-time f32 -> bf16 conversion of the 4 weight matrices into workspace.
// ---------------------------------------------------------------------------
__global__ void cvt_w_kernel(const float* __restrict__ Wq, const float* __restrict__ Wk,
                             const float* __restrict__ Wv, const float* __restrict__ Wd,
                             __bf16* __restrict__ out) {
  int tid   = blockIdx.x * 256 + threadIdx.x;   // 0 .. 262143 (float4 granules)
  int which = tid >> 16;                        // 65536 float4 per matrix
  int off   = (tid & 65535) * 4;
  const float* src = (which == 0) ? Wq : (which == 1) ? Wk : (which == 2) ? Wv : Wd;
  float4 f = *(const float4*)(src + off);
  __bf16* d = out + (size_t)which * WELEMS + off;
  d[0] = (__bf16)f.x; d[1] = (__bf16)f.y; d[2] = (__bf16)f.z; d[3] = (__bf16)f.w;
}

// ---------------------------------------------------------------------------
// Stable counting sort of concat(cluster_id, cluster_id) (values in [0,8))
// ---------------------------------------------------------------------------
__global__ void sort_kernel(const int* __restrict__ cluster_id, int* __restrict__ perm) {
  __shared__ int cnt[8];
  __shared__ int base[8];
  int t = threadIdx.x;
  if (t < 8) {
    int c = 0;
    for (int i = 0; i < N_BATCH; ++i)
      if (cluster_id[i & 1023] == t) ++c;
    cnt[t] = c;
  }
  __syncthreads();
  if (t == 0) {
    int s = 0;
    for (int v = 0; v < 8; ++v) { base[v] = s; s += cnt[v]; }
  }
  __syncthreads();
  if (t < 8) {
    int pos = base[t];
    for (int i = 0; i < N_BATCH; ++i)
      if (cluster_id[i & 1023] == t) perm[pos++] = i;
  }
}

// ---------------------------------------------------------------------------
// Fused attention kernel: one block per batch, one wave per head.
// ---------------------------------------------------------------------------
__global__ __launch_bounds__(256, 1)
void attn_kernel(const float* __restrict__ seq,
                 const float* __restrict__ mask,
                 const int*   __restrict__ perm,
                 const __bf16* __restrict__ wbf,   // Wq|Wk|Wv|Wd bf16
                 const float* __restrict__ bq, const float* __restrict__ bk,
                 const float* __restrict__ bv, const float* __restrict__ bd,
                 const float* __restrict__ ln_w, const float* __restrict__ ln_b,
                 float* __restrict__ out) {
  // LDS: [0,64K)      xb   x as bf16 [64][512]     } overlaid by hx
  //      [64K,128K)   kb   K (8 x 64x64 bf16)      } (64x512 f32) after barrier
  //      [128K,192K)  vtb  V^T (8 x 64x64 bf16)
  //      [192K,208K)  scr  per-wave 16x64 bf16
  //      [208K,272K)  ctxb ctx (64x512 bf16)
  __shared__ __align__(16) unsigned char smem[278528];
  __bf16* xb   = (__bf16*)(smem);
  __bf16* kb   = (__bf16*)(smem + 65536);
  __bf16* vtb  = (__bf16*)(smem + 131072);
  __bf16* scr  = (__bf16*)(smem + 196608);
  __bf16* ctxb = (__bf16*)(smem + 212992);
  float*  hx   = (float*)(smem);            // overlays xb+kb (128 KB)

  const int jblk = blockIdx.x;
  const int b    = perm[jblk];
  const int t    = threadIdx.x;
  const int w    = t >> 5;                  // wave id == head id
  const int lane = t & 31;
  const int n    = lane & 15;
  const int mb   = (lane >> 4) * 8;

  const float* xg    = seq + (size_t)b * CE;                // [64][512] f32
  const float* maskp = mask + (size_t)jblk * (C_DIM * C_DIM);

  const __bf16* Wqb = wbf;
  const __bf16* Wkb = wbf + WELEMS;
  const __bf16* Wvb = wbf + 2 * WELEMS;
  const __bf16* Wdb = wbf + 3 * WELEMS;

  __bf16* kbh   = kb  + w * (DH * DH);
  __bf16* vtbh  = vtb + w * (DH * DH);
  __bf16* myscr = scr + w * (16 * DH);

  // ---------------- Phase 0: x -> bf16 in LDS (converted exactly once) -----
  for (int i = t; i < CE / 4; i += 256) {
    float4 f = ((const float4*)xg)[i];
    __bf16* d = xb + i * 4;
    d[0] = (__bf16)f.x; d[1] = (__bf16)f.y; d[2] = (__bf16)f.z; d[3] = (__bf16)f.w;
  }
  __syncthreads();

  // ---------------- Phase 1: K and V projections (per head) ----------------
#pragma unroll 1
  for (int mt = 0; mt < 4; ++mt) {
    v8f kacc[4], vacc[4];
#pragma unroll
    for (int nt = 0; nt < 4; ++nt) {
      kacc[nt] = (v8f){0.f,0.f,0.f,0.f,0.f,0.f,0.f,0.f};
      vacc[nt] = (v8f){0.f,0.f,0.f,0.f,0.f,0.f,0.f,0.f};
    }
    for (int kk = 0; kk < 16; ++kk) {
      v16bf a = load_A_lds(xb + mt * 16 * E_DIM + kk * 32, E_DIM, lane);
#pragma unroll
      for (int nt = 0; nt < 4; ++nt) {
        v16bf bkm = load_B_wbf(Wkb, w * DH + nt * 16, kk * 32, lane);
        kacc[nt] = wmma_bf16(a, bkm, kacc[nt]);
        v16bf bvm = load_B_wbf(Wvb, w * DH + nt * 16, kk * 32, lane);
        vacc[nt] = wmma_bf16(a, bvm, vacc[nt]);
      }
    }
#pragma unroll
    for (int nt = 0; nt < 4; ++nt) {
      float biask = bk[w * DH + nt * 16 + n];
      float biasv = bv[w * DH + nt * 16 + n];
#pragma unroll
      for (int r = 0; r < 8; ++r) {
        int row = mt * 16 + mb + r;
        kbh[row * DH + nt * 16 + n]    = (__bf16)(kacc[nt][r] + biask); // K natural
        vtbh[(nt * 16 + n) * DH + row] = (__bf16)(vacc[nt][r] + biasv); // V^T
      }
    }
  }

  // ---------------- Phase 2: attention per 16-row Q tile --------------------
  const float scale = 0.125f;   // 1/sqrt(64)
#pragma unroll 1
  for (int mt = 0; mt < 4; ++mt) {
    // Q tile -> scratch
    {
      v8f qacc[4];
#pragma unroll
      for (int nt = 0; nt < 4; ++nt)
        qacc[nt] = (v8f){0.f,0.f,0.f,0.f,0.f,0.f,0.f,0.f};
      for (int kk = 0; kk < 16; ++kk) {
        v16bf a = load_A_lds(xb + mt * 16 * E_DIM + kk * 32, E_DIM, lane);
#pragma unroll
        for (int nt = 0; nt < 4; ++nt) {
          v16bf bm = load_B_wbf(Wqb, w * DH + nt * 16, kk * 32, lane);
          qacc[nt] = wmma_bf16(a, bm, qacc[nt]);
        }
      }
#pragma unroll
      for (int nt = 0; nt < 4; ++nt) {
        float biasq = bq[w * DH + nt * 16 + n];
#pragma unroll
        for (int r = 0; r < 8; ++r)
          myscr[(mb + r) * DH + nt * 16 + n] = (__bf16)(qacc[nt][r] + biasq);
      }
    }

    // S = Q K^T * scale + mask
    v8f s[4];
#pragma unroll
    for (int st = 0; st < 4; ++st) {
      v8f acc = {0.f,0.f,0.f,0.f,0.f,0.f,0.f,0.f};
#pragma unroll
      for (int kk = 0; kk < 2; ++kk) {
        v16bf a  = load_A_lds(myscr + kk * 32, DH, lane);
        v16bf bm = load_B_lds(kbh + (st * 16) * DH + kk * 32, DH, lane);
        acc = wmma_bf16(a, bm, acc);
      }
#pragma unroll
      for (int r = 0; r < 8; ++r)
        acc[r] = acc[r] * scale +
                 maskp[(mt * 16 + mb + r) * C_DIM + st * 16 + n];
      s[st] = acc;
    }

    // row softmax (row lives in 16 lanes of one half-wave, element r)
#pragma unroll
    for (int r = 0; r < 8; ++r) {
      float mx = fmaxf(fmaxf(s[0][r], s[1][r]), fmaxf(s[2][r], s[3][r]));
      mx = fmaxf(mx, __shfl_xor(mx, 1));
      mx = fmaxf(mx, __shfl_xor(mx, 2));
      mx = fmaxf(mx, __shfl_xor(mx, 4));
      mx = fmaxf(mx, __shfl_xor(mx, 8));
      float e0 = __expf(s[0][r] - mx);
      float e1 = __expf(s[1][r] - mx);
      float e2 = __expf(s[2][r] - mx);
      float e3 = __expf(s[3][r] - mx);
      float sm = e0 + e1 + e2 + e3;
      sm += __shfl_xor(sm, 1);
      sm += __shfl_xor(sm, 2);
      sm += __shfl_xor(sm, 4);
      sm += __shfl_xor(sm, 8);
      float inv = 1.0f / sm;
      s[0][r] = e0 * inv; s[1][r] = e1 * inv;
      s[2][r] = e2 * inv; s[3][r] = e3 * inv;
    }

    // P -> scratch (Q fully consumed)
#pragma unroll
    for (int st = 0; st < 4; ++st)
#pragma unroll
      for (int r = 0; r < 8; ++r)
        myscr[(mb + r) * DH + st * 16 + n] = (__bf16)s[st][r];

    // ctx = P V   (B[k][n] = V[k][n] = vtb[n][k], contiguous)
#pragma unroll
    for (int nt = 0; nt < 4; ++nt) {
      v8f acc = {0.f,0.f,0.f,0.f,0.f,0.f,0.f,0.f};
#pragma unroll
      for (int kk = 0; kk < 2; ++kk) {
        v16bf a  = load_A_lds(myscr + kk * 32, DH, lane);
        v16bf bm = load_B_lds(vtbh + (nt * 16) * DH + kk * 32, DH, lane);
        acc = wmma_bf16(a, bm, acc);
      }
#pragma unroll
      for (int r = 0; r < 8; ++r)
        ctxb[(mt * 16 + mb + r) * E_DIM + w * DH + nt * 16 + n] = (__bf16)acc[r];
    }
  }
  __syncthreads();

  // ---------------- Phase 3: out = ctx @ Wd^T + bd + x ----------------------
#pragma unroll 1
  for (int mt = 0; mt < 4; ++mt) {
    v8f acc[4];
#pragma unroll
    for (int nt = 0; nt < 4; ++nt)
      acc[nt] = (v8f){0.f,0.f,0.f,0.f,0.f,0.f,0.f,0.f};
    for (int kk = 0; kk < 16; ++kk) {
      v16bf a = load_A_lds(ctxb + mt * 16 * E_DIM + kk * 32, E_DIM, lane);
#pragma unroll
      for (int nt = 0; nt < 4; ++nt) {
        v16bf bm = load_B_wbf(Wdb, w * DH + nt * 16, kk * 32, lane);
        acc[nt] = wmma_bf16(a, bm, acc[nt]);
      }
    }
#pragma unroll
    for (int nt = 0; nt < 4; ++nt) {
      int col = w * DH + nt * 16 + n;
      float bias = bd[col];
#pragma unroll
      for (int r = 0; r < 8; ++r) {
        int row = mt * 16 + mb + r;
        hx[row * E_DIM + col] = acc[nt][r] + bias + xg[row * E_DIM + col];
      }
    }
  }
  __syncthreads();

  // ---------------- Phase 4: LayerNorm (4 lanes per row) --------------------
  {
    int row = t >> 2;
    int q4  = t & 3;
    const float* hrow = hx + row * E_DIM + q4 * 128;
    float sum = 0.f, sq = 0.f;
    for (int i = 0; i < 128; ++i) {
      float v = hrow[i];
      sum += v; sq += v * v;
    }
    sum += __shfl_xor(sum, 1); sum += __shfl_xor(sum, 2);
    sq  += __shfl_xor(sq, 1);  sq  += __shfl_xor(sq, 2);
    float mean = sum * (1.0f / 512.0f);
    float var  = sq * (1.0f / 512.0f) - mean * mean;
    float rstd = rsqrtf(var + 1e-12f);
    float* orow = out + (size_t)b * CE + row * E_DIM;
    for (int i = 0; i < 128; ++i) {
      int c = q4 * 128 + i;
      float v = (hrow[i] - mean) * rstd;
      orow[c] = ln_w[c] * v + ln_b[c];
    }
  }
}

// ---------------------------------------------------------------------------
extern "C" void kernel_launch(void* const* d_in, const int* in_sizes, int n_in,
                              void* d_out, int out_size, void* d_ws, size_t ws_size,
                              hipStream_t stream) {
  const float* seq        = (const float*)d_in[0];
  const float* mask       = (const float*)d_in[1];
  const int*   cluster_id = (const int*)  d_in[2];
  const float* Wq = (const float*)d_in[3];
  const float* bq = (const float*)d_in[4];
  const float* Wk = (const float*)d_in[5];
  const float* bk = (const float*)d_in[6];
  const float* Wv = (const float*)d_in[7];
  const float* bv = (const float*)d_in[8];
  const float* Wd = (const float*)d_in[9];
  const float* bd = (const float*)d_in[10];
  const float* ln_w = (const float*)d_in[11];
  const float* ln_b = (const float*)d_in[12];
  float* out = (float*)d_out;

  int*    perm = (int*)d_ws;                         // 2048 * 4 B
  __bf16* wbf  = (__bf16*)((char*)d_ws + 8192);      // 4 * 512KB bf16 weights

  cvt_w_kernel<<<1024, 256, 0, stream>>>(Wq, Wk, Wv, Wd, wbf);
  sort_kernel<<<1, 32, 0, stream>>>(cluster_id, perm);
  attn_kernel<<<N_BATCH, 256, 0, stream>>>(seq, mask, perm, wbf,
                                           bq, bk, bv, bd, ln_w, ln_b, out);
}
